// MultiHeadAttention_71339406786915
// MI455X (gfx1250) — matrix operand order repaired and checked by reading, not compiled, not run on hardware
//
#include <hip/hip_runtime.h>
#include <hip/hip_bf16.h>

typedef __attribute__((ext_vector_type(16))) _Float16 v16h;
typedef __attribute__((ext_vector_type(8)))  _Float16 v8h;
typedef __attribute__((ext_vector_type(8)))  float    v8f;
typedef __attribute__((ext_vector_type(4)))  unsigned u32x4;
typedef __attribute__((ext_vector_type(4)))  int      i32x4;
typedef __attribute__((ext_vector_type(8)))  int      i32x8;

constexpr int D_MODEL = 1024;
constexpr int D_K     = 64;
constexpr int NHEADS  = 16;
constexpr int BB      = 4;
constexpr int TT      = 2048;
constexpr int NH      = NHEADS * D_K;      // 1024
constexpr int MTOT    = BB * TT;           // 8192
constexpr float MASK_VALUE = -10000000.0f;
constexpr float QK_SCALE   = 0.125f;       // 1/sqrt(64)

#if defined(__has_builtin)
#if __has_builtin(__builtin_amdgcn_tensor_load_to_lds)
#define HAVE_TDM 1
#endif
#endif
#ifndef HAVE_TDM
#define HAVE_TDM 0
#endif

// ---------------------------------------------------------------------------
// CDNA5 async global->LDS copy (ASYNCcnt-tracked DMA), 16B per lane
// ---------------------------------------------------------------------------
__device__ __forceinline__ void async_copy_b128(unsigned lds_off, const void* gptr) {
    unsigned long long ga = (unsigned long long)(uintptr_t)gptr;
    asm volatile("global_load_async_to_lds_b128 %0, %1, off"
                 :: "v"(lds_off), "v"(ga) : "memory");
}
__device__ __forceinline__ unsigned lds_addr_of(const void* p) {
    // LDS aperture: flat addr low 32 bits are the LDS byte offset (ISA 10.2)
    return (unsigned)(uintptr_t)p;
}

#if HAVE_TDM
// ---------------------------------------------------------------------------
// TDM: DMA a 64-row x 32-element f16 tile (rows strided by 1024 elements in a
// 1024x1024 tensor) from global into contiguous LDS.  D# layout per ISA ch.8.
// ---------------------------------------------------------------------------
__device__ __forceinline__ void tdm_load_tile_64x32(const _Float16* gsrc, unsigned lds_off) {
    unsigned long long ga = (unsigned long long)(uintptr_t)gsrc;
    u32x4 g0;
    g0[0] = 1u;                                            // count=1 (valid user D#)
    g0[1] = lds_off;                                       // lds_addr (bytes)
    g0[2] = (unsigned)(ga & 0xffffffffu);                  // global_addr[31:0]
    g0[3] = (unsigned)((ga >> 32) & 0x1ffffffu) | (2u << 30); // addr[56:32] | type=2
    i32x8 g1;
    g1[0] = 0x00010000;                                    // data_size=2B; no multicast
    g1[1] = (int)(1024u << 16);                            // tensor_dim0=1024 (bits 79:48)
    g1[2] = (int)(1024u << 16);                            // tensor_dim1=1024 (bits 111:80)
    g1[3] = (int)(32u << 16);                              // tile_dim0=32 (bits 127:112)
    g1[4] = 64;                                            // tile_dim1=64, tile_dim2=0
    g1[5] = 1024;                                          // tensor_dim0_stride=1024
    g1[6] = 0;                                             // stride hi / dim1_stride lo
    g1[7] = 0;
    i32x4 gz = {0, 0, 0, 0};
#if defined(__clang_major__) && (__clang_major__ >= 23)
    i32x8 gz8 = {0, 0, 0, 0, 0, 0, 0, 0};
    __builtin_amdgcn_tensor_load_to_lds(g0, g1, gz, gz, gz8, 0);
#else
    __builtin_amdgcn_tensor_load_to_lds(g0, g1, gz, gz, 0);
#endif
}
#endif

// ---------------------------------------------------------------------------
// Weight transpose + fp32 -> f16 convert: W[k][n] (1024x1024) -> Wt[n][k]
// ---------------------------------------------------------------------------
__global__ void __launch_bounds__(256)
mha_transpose_f32_f16(const float* __restrict__ W, _Float16* __restrict__ Wt) {
    __shared__ float tile[32][33];
    const int nb = blockIdx.x * 32;
    const int kb = blockIdx.y * 32;
    const int tx = threadIdx.x;            // 0..31
    const int ty = threadIdx.y;            // 0..7
#pragma unroll
    for (int i = 0; i < 32; i += 8)
        tile[ty + i][tx] = W[(size_t)(kb + ty + i) * D_MODEL + (nb + tx)];
    __syncthreads();
#pragma unroll
    for (int i = 0; i < 32; i += 8)
        Wt[(size_t)(nb + ty + i) * D_MODEL + (kb + tx)] = (_Float16)tile[tx][ty + i];
}

// ---------------------------------------------------------------------------
// Fused QKV projection:  Y = X(fp32, 8192x1024) * Wt(f16,[n][k]) + bias
// block = 128 (4 waves); wave computes 32 rows x 64 cols.  B tile (64x32 f16)
// staged into LDS by the Tensor Data Mover (wave 0 issues, double-buffered,
// TENSORcnt-synchronized); async-DMA fallback if the TDM builtin is absent.
// blockIdx.z: 0 -> Q (b,h,t,d)   1 -> K (b,h,t,d)   2 -> V^T (b,h,d,t)
// ---------------------------------------------------------------------------
__global__ void __launch_bounds__(128)
mha_gemm_qkv(const float* __restrict__ Xq, const float* __restrict__ Xk,
             const float* __restrict__ Xv,
             const _Float16* __restrict__ Wqt, const _Float16* __restrict__ Wkt,
             const _Float16* __restrict__ Wvt,
             const float* __restrict__ bq, const float* __restrict__ bk,
             const float* __restrict__ bv,
             _Float16* __restrict__ Qh, _Float16* __restrict__ Kh,
             _Float16* __restrict__ Vth) {
    __shared__ alignas(32) _Float16 btile[2][64][32];   // [buf][n][k], 8 KB

    const int which = blockIdx.z;
    const float*    X    = (which == 0) ? Xq  : (which == 1) ? Xk  : Xv;
    const _Float16* Wt   = (which == 0) ? Wqt : (which == 1) ? Wkt : Wvt;
    const float*    bias = (which == 0) ? bq  : (which == 1) ? bk  : bv;

    const int tid  = threadIdx.x;
    const int lane = tid & 31;
    const int wave = tid >> 5;
    const int lm   = lane & 15;
    const int lh   = lane >> 4;
    const int m0   = blockIdx.y * 128 + wave * 32;
    const int n0   = blockIdx.x * 64;

    const unsigned lds_base = lds_addr_of(&btile[0][0][0]);
    const _Float16* wtile   = Wt + (size_t)n0 * D_MODEL;    // 64 rows x 1024, stride 1024

#if HAVE_TDM
    if (wave == 0) tdm_load_tile_64x32(wtile + 0, lds_base);
#else
    const int c0 = tid * 2,     row0 = c0 >> 2, seg0 = c0 & 3;
    const int c1 = tid * 2 + 1, row1 = c1 >> 2, seg1 = c1 & 3;
    async_copy_b128(lds_base + c0 * 16, wtile + (size_t)row0 * D_MODEL + seg0 * 8);
    async_copy_b128(lds_base + c1 * 16, wtile + (size_t)row1 * D_MODEL + seg1 * 8);
#endif

    v8f acc[2][4] = {};
    const float* xrow0 = X + (size_t)(m0 + lm) * D_MODEL;
    const float* xrow1 = X + (size_t)(m0 + 16 + lm) * D_MODEL;

    for (int kb = 0; kb < D_MODEL; kb += 32) {
        const int cur = (kb >> 5) & 1;
#if HAVE_TDM
        if (wave == 0) {
            if (kb + 32 < D_MODEL) {
                tdm_load_tile_64x32(wtile + kb + 32, lds_base + (cur ^ 1) * 4096);
                asm volatile("s_wait_tensorcnt 0x1" ::: "memory");   // cur tile landed
            } else {
                asm volatile("s_wait_tensorcnt 0x0" ::: "memory");
            }
        }
#else
        if (kb + 32 < D_MODEL) {
            const unsigned nb = lds_base + (cur ^ 1) * 4096;
            async_copy_b128(nb + c0 * 16, wtile + (size_t)row0 * D_MODEL + kb + 32 + seg0 * 8);
            async_copy_b128(nb + c1 * 16, wtile + (size_t)row1 * D_MODEL + kb + 32 + seg1 * 8);
            asm volatile("s_wait_asynccnt 0x2" ::: "memory");
        } else {
            asm volatile("s_wait_asynccnt 0x0" ::: "memory");
        }
#endif
        __syncthreads();                                    // all waves see cur buffer

        // A fragments (fp32 -> f16), two 16-row tiles
        v16h a[2];
#pragma unroll
        for (int mt = 0; mt < 2; ++mt) {
            const float* xr = mt ? xrow1 : xrow0;
            alignas(16) float f[16];
            const float4* p0 = reinterpret_cast<const float4*>(xr + kb + lh * 16);
            const float4* p1 = reinterpret_cast<const float4*>(xr + kb + 8 + lh * 16);
            reinterpret_cast<float4*>(f)[0] = p0[0];
            reinterpret_cast<float4*>(f)[1] = p0[1];
            reinterpret_cast<float4*>(f)[2] = p1[0];
            reinterpret_cast<float4*>(f)[3] = p1[1];
#pragma unroll
            for (int e = 0; e < 16; ++e) a[mt][e] = (_Float16)f[e];
        }

        // preload all B fragments, then 8 back-to-back WMMAs
        v16h bfr[4];
#pragma unroll
        for (int nt = 0; nt < 4; ++nt)
            bfr[nt] = *reinterpret_cast<const v16h*>(&btile[cur][nt * 16 + lm][lh * 16]);
#pragma unroll
        for (int nt = 0; nt < 4; ++nt) {
            acc[0][nt] = __builtin_amdgcn_wmma_f32_16x16x32_f16(
                false, a[0], false, bfr[nt], (short)0, acc[0][nt], false, false);
            acc[1][nt] = __builtin_amdgcn_wmma_f32_16x16x32_f16(
                false, a[1], false, bfr[nt], (short)0, acc[1][nt], false, false);
        }
        __syncthreads();                                    // done reading cur buffer
    }

    // Epilogue: bias add, f16 convert, scatter into attention layouts
#pragma unroll
    for (int nt = 0; nt < 4; ++nt) {
        const int   n   = n0 + nt * 16 + lm;
        const float bv_ = bias[n];
        const int   h   = n >> 6;
        const int   d   = n & 63;
#pragma unroll
        for (int mt = 0; mt < 2; ++mt) {
#pragma unroll
            for (int r = 0; r < 8; ++r) {
                const int m = m0 + mt * 16 + lh * 8 + r;    // global row
                const int b = m / TT;
                const int t = m % TT;
                const _Float16 hv = (_Float16)(acc[mt][nt][r] + bv_);
                if (which == 0)
                    Qh[(((size_t)b * NHEADS + h) * TT + t) * D_K + d] = hv;
                else if (which == 1)
                    Kh[(((size_t)b * NHEADS + h) * TT + t) * D_K + d] = hv;
                else
                    Vth[(((size_t)b * NHEADS + h) * D_K + d) * TT + t] = hv;
            }
        }
    }
}

// ---------------------------------------------------------------------------
// Flash attention: one wave = two 16-query tiles (32 rows) of one (b,h).
// grid = (T/128, B*H), block = 128 (4 waves).
// ---------------------------------------------------------------------------
__global__ void __launch_bounds__(128)
mha_attention(const _Float16* __restrict__ Qh, const _Float16* __restrict__ Kh,
              const _Float16* __restrict__ Vth, const int* __restrict__ valid_len,
              _Float16* __restrict__ AOh) {
    __shared__ alignas(32) _Float16 pbuf[4][2][16][40];   // per-wave P tiles, padded rows

    const int lane = threadIdx.x & 31;
    const int wave = threadIdx.x >> 5;
    const int lm   = lane & 15;
    const int lh   = lane >> 4;
    const int bh   = blockIdx.y;
    const int b    = bh >> 4;
    const int h    = bh & 15;
    const int qt0  = blockIdx.x * 128 + wave * 32;
    const int valid = valid_len[b];

    // Q A-fragments (two m-tiles x two 32-wide k-steps), pre-scaled by 1/sqrt(dk)
    v16h aq[2][2];
#pragma unroll
    for (int mt = 0; mt < 2; ++mt) {
        const _Float16* qrow = Qh + ((size_t)bh * TT + qt0 + mt * 16 + lm) * D_K;
#pragma unroll
        for (int kk = 0; kk < 2; ++kk) {
            const v8h r0 = *reinterpret_cast<const v8h*>(qrow + kk * 32 + lh * 16);
            const v8h r1 = *reinterpret_cast<const v8h*>(qrow + kk * 32 + 8 + lh * 16);
#pragma unroll
            for (int e = 0; e < 8; ++e) {
                aq[mt][kk][e]     = r0[e] * (_Float16)QK_SCALE;
                aq[mt][kk][8 + e] = r1[e] * (_Float16)QK_SCALE;
            }
        }
    }

    v8f   o[2][4] = {};
    float rmax[2][8], rsum[2][8];
#pragma unroll
    for (int mt = 0; mt < 2; ++mt)
#pragma unroll
        for (int r = 0; r < 8; ++r) { rmax[mt][r] = -3.0e38f; rsum[mt][r] = 0.0f; }

    const _Float16* Kbase = Kh  + (size_t)bh * TT * D_K;
    const _Float16* Vbase = Vth + (size_t)bh * D_K * TT;

    for (int kt = 0; kt < TT; kt += 32) {
        // preload all 4 K fragments, then 8 back-to-back WMMAs for S = Q*K^T
        v16h kfr[2][2];
#pragma unroll
        for (int c = 0; c < 2; ++c) {
            const _Float16* krow = Kbase + (size_t)(kt + c * 16 + lm) * D_K;
#pragma unroll
            for (int kk = 0; kk < 2; ++kk)
                kfr[c][kk] = *reinterpret_cast<const v16h*>(krow + kk * 32 + lh * 16);
        }
        v8f s[2][2] = {};
#pragma unroll
        for (int c = 0; c < 2; ++c)
#pragma unroll
            for (int kk = 0; kk < 2; ++kk) {
                s[0][c] = __builtin_amdgcn_wmma_f32_16x16x32_f16(
                    false, aq[0][kk], false, kfr[c][kk], (short)0, s[0][c], false, false);
                s[1][c] = __builtin_amdgcn_wmma_f32_16x16x32_f16(
                    false, aq[1][kk], false, kfr[c][kk], (short)0, s[1][c], false, false);
            }

        // key-padding mask + online softmax per m-tile
#pragma unroll
        for (int mt = 0; mt < 2; ++mt) {
            float tmax[8];
#pragma unroll
            for (int r = 0; r < 8; ++r) tmax[r] = -3.0e38f;
#pragma unroll
            for (int c = 0; c < 2; ++c) {
                const bool masked = (kt + c * 16 + lm) >= valid;
#pragma unroll
                for (int r = 0; r < 8; ++r) {
                    const float sv = masked ? MASK_VALUE : s[mt][c][r];
                    s[mt][c][r] = sv;
                    tmax[r] = fmaxf(tmax[r], sv);
                }
            }
#pragma unroll
            for (int r = 0; r < 8; ++r) {
                float v = tmax[r];
                v = fmaxf(v, __shfl_xor(v, 1));
                v = fmaxf(v, __shfl_xor(v, 2));
                v = fmaxf(v, __shfl_xor(v, 4));
                v = fmaxf(v, __shfl_xor(v, 8));
                const float nm  = fmaxf(rmax[mt][r], v);
                const float fac = __expf(rmax[mt][r] - nm);
                rmax[mt][r] = nm;
                rsum[mt][r] *= fac;
#pragma unroll
                for (int dt = 0; dt < 4; ++dt) o[mt][dt][r] *= fac;
            }
            // P = exp(S - m); per-lane partial row sums; stage into LDS (C->A layout)
#pragma unroll
            for (int c = 0; c < 2; ++c) {
#pragma unroll
                for (int r = 0; r < 8; ++r) {
                    const float p = __expf(s[mt][c][r] - rmax[mt][r]);
                    rsum[mt][r] += p;
                    pbuf[wave][mt][lh * 8 + r][c * 16 + lm] = (_Float16)p;
                }
            }
        }
        asm volatile("s_wait_dscnt 0x0" ::: "memory");

        // reload both P tiles as A fragments
        v16h pa[2];
#pragma unroll
        for (int mt = 0; mt < 2; ++mt) {
            const _Float16* prow = &pbuf[wave][mt][lm][0];
            const v8h p0 = *reinterpret_cast<const v8h*>(prow + lh * 16);
            const v8h p1 = *reinterpret_cast<const v8h*>(prow + 8 + lh * 16);
#pragma unroll
            for (int e = 0; e < 8; ++e) { pa[mt][e] = p0[e]; pa[mt][8 + e] = p1[e]; }
        }

        // preload all 4 V fragments, then 8 WMMAs for O += P * V
        v16h vfr[4];
#pragma unroll
        for (int dt = 0; dt < 4; ++dt)
            vfr[dt] = *reinterpret_cast<const v16h*>(
                Vbase + (size_t)(dt * 16 + lm) * TT + kt + lh * 16);
#pragma unroll
        for (int dt = 0; dt < 4; ++dt) {
            o[0][dt] = __builtin_amdgcn_wmma_f32_16x16x32_f16(
                false, pa[0], false, vfr[dt], (short)0, o[0][dt], false, false);
            o[1][dt] = __builtin_amdgcn_wmma_f32_16x16x32_f16(
                false, pa[1], false, vfr[dt], (short)0, o[1][dt], false, false);
        }
    }

    // finalize: reduce row sums across lanes, normalize, store (b, t, h*64+d)
#pragma unroll
    for (int mt = 0; mt < 2; ++mt) {
        float inv[8];
#pragma unroll
        for (int r = 0; r < 8; ++r) {
            float v = rsum[mt][r];
            v += __shfl_xor(v, 1);
            v += __shfl_xor(v, 2);
            v += __shfl_xor(v, 4);
            v += __shfl_xor(v, 8);
            inv[r] = 1.0f / v;
        }
#pragma unroll
        for (int dt = 0; dt < 4; ++dt) {
            const int n = h * D_K + dt * 16 + lm;
#pragma unroll
            for (int r = 0; r < 8; ++r) {
                const int t = qt0 + mt * 16 + lh * 8 + r;
                AOh[((size_t)b * TT + t) * NH + n] = (_Float16)(o[mt][dt][r] * inv[r]);
            }
        }
    }
}

// ---------------------------------------------------------------------------
// Output projection: Y(fp32) = AO(f16, 8192x1024) * Wot(f16,[n][k]) + b_o
// M=32/wave; B tile staged by lane-granular async DMA (ASYNCcnt path).
// ---------------------------------------------------------------------------
__global__ void __launch_bounds__(128)
mha_gemm_out(const _Float16* __restrict__ A, const _Float16* __restrict__ Wot,
             const float* __restrict__ bo, float* __restrict__ Y) {
    __shared__ alignas(32) _Float16 btile[2][64][32];

    const int tid  = threadIdx.x;
    const int lane = tid & 31;
    const int wave = tid >> 5;
    const int lm   = lane & 15;
    const int lh   = lane >> 4;
    const int m0   = blockIdx.y * 128 + wave * 32;
    const int n0   = blockIdx.x * 64;

    const unsigned lds_base = lds_addr_of(&btile[0][0][0]);
    const int c0 = tid * 2,     row0 = c0 >> 2, seg0 = c0 & 3;
    const int c1 = tid * 2 + 1, row1 = c1 >> 2, seg1 = c1 & 3;

    async_copy_b128(lds_base + c0 * 16, Wot + (size_t)(n0 + row0) * NH + 0 + seg0 * 8);
    async_copy_b128(lds_base + c1 * 16, Wot + (size_t)(n0 + row1) * NH + 0 + seg1 * 8);

    v8f acc[2][4] = {};
    const _Float16* arow0 = A + (size_t)(m0 + lm) * NH;
    const _Float16* arow1 = A + (size_t)(m0 + 16 + lm) * NH;

    for (int kb = 0; kb < NH; kb += 32) {
        const int cur = (kb >> 5) & 1;
        if (kb + 32 < NH) {
            const unsigned nb = lds_base + (cur ^ 1) * 4096;
            async_copy_b128(nb + c0 * 16, Wot + (size_t)(n0 + row0) * NH + kb + 32 + seg0 * 8);
            async_copy_b128(nb + c1 * 16, Wot + (size_t)(n0 + row1) * NH + kb + 32 + seg1 * 8);
            asm volatile("s_wait_asynccnt 0x2" ::: "memory");
        } else {
            asm volatile("s_wait_asynccnt 0x0" ::: "memory");
        }
        __syncthreads();

        v16h a[2];
#pragma unroll
        for (int mt = 0; mt < 2; ++mt) {
            const _Float16* ar = mt ? arow1 : arow0;
            const v8h a0 = *reinterpret_cast<const v8h*>(ar + kb + lh * 16);
            const v8h a1 = *reinterpret_cast<const v8h*>(ar + kb + 8 + lh * 16);
#pragma unroll
            for (int e = 0; e < 8; ++e) { a[mt][e] = a0[e]; a[mt][8 + e] = a1[e]; }
        }

        v16h bfr[4];
#pragma unroll
        for (int nt = 0; nt < 4; ++nt)
            bfr[nt] = *reinterpret_cast<const v16h*>(&btile[cur][nt * 16 + lm][lh * 16]);
#pragma unroll
        for (int nt = 0; nt < 4; ++nt) {
            acc[0][nt] = __builtin_amdgcn_wmma_f32_16x16x32_f16(
                false, a[0], false, bfr[nt], (short)0, acc[0][nt], false, false);
            acc[1][nt] = __builtin_amdgcn_wmma_f32_16x16x32_f16(
                false, a[1], false, bfr[nt], (short)0, acc[1][nt], false, false);
        }
        __syncthreads();
    }

#pragma unroll
    for (int nt = 0; nt < 4; ++nt) {
        const int   n  = n0 + nt * 16 + lm;
        const float bb = bo[n];
#pragma unroll
        for (int mt = 0; mt < 2; ++mt) {
#pragma unroll
            for (int r = 0; r < 8; ++r) {
                const int m = m0 + mt * 16 + lh * 8 + r;
                Y[(size_t)m * D_MODEL + n] = acc[mt][nt][r] + bb;
            }
        }
    }
}

// ---------------------------------------------------------------------------
// host launcher
// ---------------------------------------------------------------------------
extern "C" void kernel_launch(void* const* d_in, const int* in_sizes, int n_in,
                              void* d_out, int out_size, void* d_ws, size_t ws_size,
                              hipStream_t stream) {
    (void)in_sizes; (void)n_in; (void)out_size; (void)ws_size;

    const float* q  = (const float*)d_in[0];
    const float* k  = (const float*)d_in[1];
    const float* v  = (const float*)d_in[2];
    const int*   vl = (const int*)  d_in[3];
    const float* Wq = (const float*)d_in[4];
    const float* bq = (const float*)d_in[5];
    const float* Wk = (const float*)d_in[6];
    const float* bk = (const float*)d_in[7];
    const float* Wv = (const float*)d_in[8];
    const float* bv = (const float*)d_in[9];
    const float* Wo = (const float*)d_in[10];
    const float* bo = (const float*)d_in[11];
    float* out = (float*)d_out;

    char* ws = (char*)d_ws;
    const size_t WT_BYTES  = (size_t)D_MODEL * NH * sizeof(_Float16);           // 2 MB
    const size_t ACT_BYTES = (size_t)BB * NHEADS * TT * D_K * sizeof(_Float16); // 16 MB
    _Float16* Wqt = (_Float16*)(ws + 0 * WT_BYTES);
    _Float16* Wkt = (_Float16*)(ws + 1 * WT_BYTES);
    _Float16* Wvt = (_Float16*)(ws + 2 * WT_BYTES);
    _Float16* Wot = (_Float16*)(ws + 3 * WT_BYTES);
    _Float16* Qh  = (_Float16*)(ws + 4 * WT_BYTES + 0 * ACT_BYTES);
    _Float16* Kh  = (_Float16*)(ws + 4 * WT_BYTES + 1 * ACT_BYTES);
    _Float16* Vth = (_Float16*)(ws + 4 * WT_BYTES + 2 * ACT_BYTES);
    _Float16* AOh = (_Float16*)(ws + 4 * WT_BYTES + 3 * ACT_BYTES);

    const dim3 tpose_blk(32, 8, 1), tpose_grd(32, 32, 1);
    mha_transpose_f32_f16<<<tpose_grd, tpose_blk, 0, stream>>>(Wq, Wqt);
    mha_transpose_f32_f16<<<tpose_grd, tpose_blk, 0, stream>>>(Wk, Wkt);
    mha_transpose_f32_f16<<<tpose_grd, tpose_blk, 0, stream>>>(Wv, Wvt);
    mha_transpose_f32_f16<<<tpose_grd, tpose_blk, 0, stream>>>(Wo, Wot);

    mha_gemm_qkv<<<dim3(NH / 64, MTOT / 128, 3), 128, 0, stream>>>(
        q, k, v, Wqt, Wkt, Wvt, bq, bk, bv, Qh, Kh, Vth);

    mha_attention<<<dim3(TT / 128, BB * NHEADS, 1), 128, 0, stream>>>(
        Qh, Kh, Vth, vl, AOh);

    mha_gemm_out<<<dim3(D_MODEL / 64, MTOT / 128, 1), 128, 0, stream>>>(
        AOh, Wot, bo, out);
}
